// LMGUDDPMODFlowPredictor_63737314672932
// MI455X (gfx1250) — compile-verified, matrix operand order checked by reading, not run on hardware
//
#include <hip/hip_runtime.h>
#include <hip/hip_bf16.h>

typedef __attribute__((ext_vector_type(16))) _Float16 v16h;
typedef __attribute__((ext_vector_type(8)))  _Float16 v8h;
typedef __attribute__((ext_vector_type(8)))  float    v8f;

union FragU { v16h v; v8h h[2]; };

__device__ __forceinline__ v8f wmma_f16(const FragU& a, const FragU& b, v8f c) {
  return __builtin_amdgcn_wmma_f32_16x16x32_f16(false, a.v, false, b.v, (short)0, c, false, false);
}

__device__ __forceinline__ float fast_rcp(float x)  { return __builtin_amdgcn_rcpf(x); }
__device__ __forceinline__ float fast_silu(float v) { return v * fast_rcp(1.f + __expf(-v)); }
// tanh(x) = 1 - 2/(e^{2x}+1); saturates correctly at +-1
__device__ __forceinline__ float fast_tanh(float x) {
  return 1.f - 2.f * fast_rcp(1.f + __expf(2.f * x));
}

// ---------------------------------------------------------------------------
// Column-strip WMMA GEMM for M=128: each wave owns a 16-col strip with 8
// accumulators (one per 16-row tile). Per k-step the B-frag is loaded ONCE and
// 8 WMMAs issue back-to-back on independent accumulators (no D-reg RAW => no
// hazard NOPs, 8x less B traffic).
// A: f16 row-major [128 x K], stride lda. B: f16 [N][K] (row n = output col n).
template <int ACT, bool HB, bool HC32, bool HC16, bool HCT>
__device__ __forceinline__ void gemm_strip(
    const _Float16* A, int lda, const _Float16* B, int ldb,
    const float* bias, float scale,
    float* C32, int ldc, _Float16* C16, int ldc16,
    _Float16* CT16, int ldt, int ctmin,
    int N, int Kdim)
{
  const int lane = threadIdx.x & 31;
  const int wave = threadIdx.x >> 5;
  const int m15  = lane & 15;
  const int hsel = lane >> 4;
  for (int tc = wave; tc < (N >> 4); tc += 8) {
    v8f acc[8];
#pragma unroll
    for (int r = 0; r < 8; ++r) acc[r] = (v8f){0.f,0.f,0.f,0.f,0.f,0.f,0.f,0.f};
    const _Float16* bp = B + (tc * 16 + m15) * ldb + 8 * hsel;
    const _Float16* ap = A + m15 * lda + 8 * hsel;
    for (int k0 = 0; k0 < Kdim; k0 += 32) {
      FragU bf;
      bf.h[0] = *(const v8h*)(bp);  bf.h[1] = *(const v8h*)(bp + 16);
      bp += 32;
#pragma unroll
      for (int tr = 0; tr < 8; ++tr) {
        FragU af;
        const _Float16* a0 = ap + tr * 16 * lda + k0;
        af.h[0] = *(const v8h*)(a0);  af.h[1] = *(const v8h*)(a0 + 16);
        acc[tr] = wmma_f16(af, bf, acc[tr]);
      }
    }
    const int col = tc * 16 + m15;
    float bv = 0.f;
    if (HB) bv = bias[col];
#pragma unroll
    for (int tr = 0; tr < 8; ++tr) {
#pragma unroll
      for (int r = 0; r < 8; ++r) {
        const int row = tr * 16 + r + 8 * hsel;
        float v = acc[tr][r] * scale + bv;
        if (ACT == 1) v = v > 0.f ? v : 0.f;
        else if (ACT == 2) v = fast_silu(v);
        if (HC32) C32[row * ldc + col] = v;
        if (HC16) C16[row * ldc16 + col] = (_Float16)v;
        if (HCT) { if (col >= ctmin) CT16[(col - ctmin) * ldt + row] = (_Float16)v; }
      }
    }
  }
}

// Per-tile WMMA GEMM (used only for P@V where N=32 gives too few strips).
template <bool HC16>
__device__ __forceinline__ void gemm_tile(
    const _Float16* A, int lda, const _Float16* B, int ldb,
    _Float16* C16, int ldc16, int M, int N, int Kdim)
{
  const int lane = threadIdx.x & 31;
  const int wave = threadIdx.x >> 5;
  const int m15  = lane & 15;
  const int hsel = lane >> 4;
  const int mt = M >> 4, nt = N >> 4;
  for (int t = wave; t < mt * nt; t += 8) {
    const int tr = t % mt, tc = t / mt;
    v8f acc = {0.f,0.f,0.f,0.f,0.f,0.f,0.f,0.f};
    const _Float16* ap = A + (tr * 16 + m15) * lda + 8 * hsel;
    const _Float16* bp = B + (tc * 16 + m15) * ldb + 8 * hsel;
    for (int k0 = 0; k0 < Kdim; k0 += 32) {
      FragU a, b;
      a.h[0] = *(const v8h*)(ap);  a.h[1] = *(const v8h*)(ap + 16);
      b.h[0] = *(const v8h*)(bp);  b.h[1] = *(const v8h*)(bp + 16);
      ap += 32; bp += 32;
      acc = wmma_f16(a, b, acc);
    }
    const int col = tc * 16 + m15;
#pragma unroll
    for (int r = 0; r < 8; ++r) {
      const int row = tr * 16 + r + 8 * hsel;
      if (HC16) C16[row * ldc16 + col] = (_Float16)acc[r];
    }
  }
}

// ---------------- prep kernels: f32 -> f16 transposed weights ----------------
__global__ void transpose_f16_kernel(const float* __restrict__ src, _Float16* __restrict__ dst,
                                     int K, int N) {
  int idx = blockIdx.x * 256 + threadIdx.x;
  if (idx < K * N) { int k = idx / N, n = idx % N; dst[n * K + k] = (_Float16)src[idx]; }
}

// wcat[640][256]: rows 0..511 = s_w1[k][2+kk][d] (n = k*128+d); rows 512..639 = mw_w1[kk][n-512]
__global__ void wcat_kernel(const float* __restrict__ s_w1, const float* __restrict__ mw_w1,
                            _Float16* __restrict__ dst) {
  int idx = blockIdx.x * 256 + threadIdx.x;
  if (idx < 640 * 256) {
    int n = idx / 256, kk = idx % 256;
    float v = (n < 512) ? s_w1[(n >> 7) * 258 * 128 + (2 + kk) * 128 + (n & 127)]
                        : mw_w1[kk * 128 + (n - 512)];
    dst[idx] = (_Float16)v;
  }
}

// ---------------- kernel 1: time embedding -> combined[:, 0:128] ----------------
__global__ void __launch_bounds__(128) timeemb_kernel(
    const int* __restrict__ timesteps,
    const float* __restrict__ t_w1, const float* __restrict__ t_b1,
    const float* __restrict__ t_w2, const float* __restrict__ t_b2,
    _Float16* __restrict__ comb16) {
  __shared__ float te[128];
  __shared__ float hid[256];
  const int b = blockIdx.x, t = threadIdx.x;
  if (t < 64) {
    float f = __expf(-logf(10000.f) * (float)t / 64.f);
    float arg = (float)timesteps[b] * f;
    te[t] = cosf(arg); te[t + 64] = sinf(arg);
  }
  __syncthreads();
  for (int j = t; j < 256; j += 128) {
    float a = t_b1[j];
    for (int i = 0; i < 128; ++i) a += te[i] * t_w1[i * 256 + j];
    hid[j] = fast_silu(a);
  }
  __syncthreads();
  float a = t_b2[t];
  for (int i = 0; i < 256; ++i) a += hid[i] * t_w2[i * 128 + t];
  comb16[(long)b * 256 + t] = (_Float16)a;
}

// ---------------- kernel 2: cond MLP + 2-layer transformer, per batch element ----------------
__global__ void __launch_bounds__(256) transformer_kernel(
    const float* __restrict__ cond,              // [B,128,6]
    const float* __restrict__ c_w1, const float* __restrict__ c_b1,
    const float* __restrict__ c_b2,
    const _Float16* __restrict__ c_w2t,          // [128][64]
    const _Float16* __restrict__ wqkv_t,         // [2][384][128]
    const float* __restrict__ bqkv,              // [2][384]
    const _Float16* __restrict__ wo_t,           // [2][128][128]
    const float* __restrict__ bo,                // [2][128]
    const float* __restrict__ ln1_g, const float* __restrict__ ln1_b,
    const _Float16* __restrict__ ffw1_t,         // [2][256][128]
    const float* __restrict__ ff_b1,             // [2][256]
    const _Float16* __restrict__ ffw2_t,         // [2][128][256]
    const float* __restrict__ ff_b2,             // [2][128]
    const float* __restrict__ ln2_g, const float* __restrict__ ln2_b,
    _Float16* __restrict__ comb16)               // [B][256]
{
  extern __shared__ char smem[];
  float*    hf32  = (float*)(smem);                      // 65536: h (fp32 master)
  float*    sc32  = (float*)(smem + 65536);              // 65536: scores / gemm f32 out
  _Float16* buf16 = (_Float16*)(smem + 131072);          // 98304: qkv f16 / ffn1 f16 / probs
  _Float16* hf16  = (_Float16*)(smem + 229376);          // 32768: h f16 (reused as attn-out f16)
  _Float16* vt16  = (_Float16*)(smem + 262144);          // 32768: V^T f16 [feat][token]
  const int b = blockIdx.x;
  const int t = threadIdx.x;

  // cond MLP layer 1: (6->64) silu, f16 into buf16 stride 64 (2 threads per row)
  {
    const int row = t & 127, half = t >> 7;
    const float* cp = cond + ((long)b * 128 + row) * 6;
    float ci[6];
#pragma unroll
    for (int i = 0; i < 6; ++i) ci[i] = cp[i];
    for (int j = half * 32; j < half * 32 + 32; ++j) {
      float a = c_b1[j];
#pragma unroll
      for (int i = 0; i < 6; ++i) a += ci[i] * c_w1[i * 64 + j];
      buf16[row * 64 + j] = (_Float16)fast_silu(a);
    }
  }
  __syncthreads();
  // cond MLP layer 2: (64->128) silu -> hf32 + hf16 (WMMA)
  gemm_strip<2, true, true, true, false>(buf16, 64, c_w2t, 64, c_b2, 1.f,
                                         hf32, 128, hf16, 128, nullptr, 0, 0, 128, 64);
  __syncthreads();

  for (int l = 0; l < 2; ++l) {
    // QKV: [128,128] @ [128,384]; f16 -> buf16 (stride 384); V also transposed -> vt16
    gemm_strip<0, true, false, true, true>(hf16, 128, wqkv_t + l * 384 * 128, 128,
                                           bqkv + l * 384, 1.f,
                                           nullptr, 0, buf16, 384, vt16, 128, 256, 384, 128);
    __syncthreads();

    for (int hd = 0; hd < 4; ++hd) {
      // scores = Q @ K^T / sqrt(32); B-frags (K rows) read once per 16-col strip
      gemm_strip<0, false, true, false, false>(buf16 + hd * 32, 384, buf16 + 128 + hd * 32, 384,
                                               nullptr, 0.1767766953f,
                                               sc32, 128, nullptr, 0, nullptr, 0, 0, 128, 32);
      __syncthreads();
      // row softmax -> probs f16 in buf16[:,256:384]
      if (t < 128) {
        float* sr = sc32 + t * 128;
        float mx = sr[0];
        for (int j = 1; j < 128; ++j) mx = fmaxf(mx, sr[j]);
        float sum = 0.f;
        for (int j = 0; j < 128; ++j) { float e = __expf(sr[j] - mx); sr[j] = e; sum += e; }
        float inv = fast_rcp(sum);
        _Float16* pr = buf16 + t * 384 + 256;
        for (int j = 0; j < 128; ++j) pr[j] = (_Float16)(sr[j] * inv);
      }
      __syncthreads();
      // o_head = P @ V (V^T gives contiguous B-frags); f16 -> hf16[:, hd*32..]
      gemm_tile<true>(buf16 + 256, 384, vt16 + hd * 32 * 128, 128,
                      hf16 + hd * 32, 128, 128, 32, 128);
      __syncthreads();
    }
    // output projection -> sc32
    gemm_strip<0, true, true, false, false>(hf16, 128, wo_t + l * 128 * 128, 128, bo + l * 128,
                                            1.f, sc32, 128, nullptr, 0, nullptr, 0, 0, 128, 128);
    __syncthreads();
    // residual + LN1 -> hf32, hf16
    if (t < 128) {
      float* hr = hf32 + t * 128; const float* orow = sc32 + t * 128;
      float m = 0.f;
      for (int j = 0; j < 128; ++j) { float x = hr[j] + orow[j]; hr[j] = x; m += x; }
      m *= (1.f / 128.f);
      float v = 0.f;
      for (int j = 0; j < 128; ++j) { float d = hr[j] - m; v += d * d; }
      v *= (1.f / 128.f);
      float inv = rsqrtf(v + 1e-5f);
      for (int j = 0; j < 128; ++j) {
        float y = ln1_g[l * 128 + j] * (hr[j] - m) * inv + ln1_b[l * 128 + j];
        hr[j] = y; hf16[t * 128 + j] = (_Float16)y;
      }
    }
    __syncthreads();
    // FFN1 (128->256) relu, f16 -> buf16 stride 256
    gemm_strip<1, true, false, true, false>(hf16, 128, ffw1_t + l * 256 * 128, 128,
                                            ff_b1 + l * 256, 1.f,
                                            nullptr, 0, buf16, 256, nullptr, 0, 0, 256, 128);
    __syncthreads();
    // FFN2 (256->128) -> sc32
    gemm_strip<0, true, true, false, false>(buf16, 256, ffw2_t + l * 128 * 256, 256,
                                            ff_b2 + l * 128, 1.f,
                                            sc32, 128, nullptr, 0, nullptr, 0, 0, 128, 256);
    __syncthreads();
    // residual + LN2 -> hf32, hf16
    if (t < 128) {
      float* hr = hf32 + t * 128; const float* frow = sc32 + t * 128;
      float m = 0.f;
      for (int j = 0; j < 128; ++j) { float x = hr[j] + frow[j]; hr[j] = x; m += x; }
      m *= (1.f / 128.f);
      float v = 0.f;
      for (int j = 0; j < 128; ++j) { float d = hr[j] - m; v += d * d; }
      v *= (1.f / 128.f);
      float inv = rsqrtf(v + 1e-5f);
      for (int j = 0; j < 128; ++j) {
        float y = ln2_g[l * 128 + j] * (hr[j] - m) * inv + ln2_b[l * 128 + j];
        hr[j] = y; hf16[t * 128 + j] = (_Float16)y;
      }
    }
    __syncthreads();
  }
  // cond_emb = mean over S -> combined[:, 128:256]
  if (t < 128) {
    float s = 0.f;
    for (int r = 0; r < 128; ++r) s += hf32[r * 128 + t];
    comb16[(long)b * 256 + 128 + t] = (_Float16)(s * (1.f / 128.f));
  }
}

// ---------------- kernel 3: Y[4096,640] = combined @ [s_w1_c | mw_w1] (WMMA, strip) ----------
// Each wave: one 16-col strip x 128 rows (8 row tiles, 8 accumulators).
__global__ void __launch_bounds__(256) base_gemm_kernel(
    const _Float16* __restrict__ comb16,   // [4096][256]
    const _Float16* __restrict__ wcat,     // [640][256]
    const float* __restrict__ s_b1,        // [512]
    const float* __restrict__ mw_b1,       // [128]
    float* __restrict__ basebuf,           // [4096][512]
    float* __restrict__ zbuf)              // [4096][128]
{
  const int lane = threadIdx.x & 31, wave = threadIdx.x >> 5;
  const int task = blockIdx.x * 8 + wave;          // 32 row-strips * 40 col-tiles = 1280 tasks
  const int rs = task % 32, tc = task / 32;
  const int m15 = lane & 15, hsel = lane >> 4;
  const _Float16* ap0 = comb16 + ((long)(rs * 128 + m15)) * 256 + 8 * hsel;
  const _Float16* bp  = wcat + (tc * 16 + m15) * 256 + 8 * hsel;
  v8f acc[8];
#pragma unroll
  for (int r = 0; r < 8; ++r) acc[r] = (v8f){0.f,0.f,0.f,0.f,0.f,0.f,0.f,0.f};
  for (int k0 = 0; k0 < 256; k0 += 32) {
    FragU bf;
    bf.h[0] = *(const v8h*)(bp);  bf.h[1] = *(const v8h*)(bp + 16);
    bp += 32;
#pragma unroll
    for (int tr = 0; tr < 8; ++tr) {
      FragU af;
      const _Float16* a0 = ap0 + tr * 16 * 256 + k0;
      af.h[0] = *(const v8h*)(a0);  af.h[1] = *(const v8h*)(a0 + 16);
      acc[tr] = wmma_f16(af, bf, acc[tr]);
    }
  }
  const int col = tc * 16 + m15;
#pragma unroll
  for (int tr = 0; tr < 8; ++tr) {
#pragma unroll
    for (int r = 0; r < 8; ++r) {
      const long row = rs * 128 + tr * 16 + r + 8 * hsel;
      float v = acc[tr][r];
      if (col < 512) basebuf[row * 512 + col] = v + s_b1[col];
      else           zbuf[row * 128 + (col - 512)] = v + mw_b1[col - 512];
    }
  }
}

// ---------------- kernel 3b: mixture weights softmax ----------------
__global__ void __launch_bounds__(128) mix_kernel(
    const float* __restrict__ zbuf, const float* __restrict__ mw_w2,
    const float* __restrict__ mw_b2, float* __restrict__ mwbuf) {
  __shared__ float sm[128];
  __shared__ float lg[4];
  const int b = blockIdx.x, t = threadIdx.x;
  sm[t] = fast_silu(zbuf[(long)b * 128 + t]);
  __syncthreads();
  if (t < 4) {
    float a = mw_b2[t];
    for (int d = 0; d < 128; ++d) a += sm[d] * mw_w2[d * 4 + t];
    lg[t] = a;
  }
  __syncthreads();
  if (t == 0) {
    float mx = fmaxf(fmaxf(lg[0], lg[1]), fmaxf(lg[2], lg[3]));
    float e[4], s = 0.f;
    for (int k = 0; k < 4; ++k) { e[k] = __expf(lg[k] - mx); s += e[k]; }
    float inv = fast_rcp(s);
    for (int k = 0; k < 4; ++k) mwbuf[(long)b * 4 + k] = e[k] * inv;
  }
}

// ---------------- kernel 4: students (rank-2 correction + tanh + mixture) ----------------
__global__ void __launch_bounds__(128) student_kernel(
    const float* __restrict__ x,        // [B,2,128]
    const float* __restrict__ basebuf,  // [B,512]
    const float* __restrict__ mwbuf,    // [B,4]
    const float* __restrict__ s_w1,     // [4,258,128]
    const float* __restrict__ s_w2,     // [4,128,2]
    const float* __restrict__ s_b2,     // [4,2]
    float* __restrict__ out)            // [B,2,128]
{
  __shared__ float base[512], c0[512], c1[512], w0[512], w1[512];
  __shared__ float bc[2];
  const int b = blockIdx.x, t = threadIdx.x;
  float mw[4];
#pragma unroll
  for (int k = 0; k < 4; ++k) mw[k] = mwbuf[(long)b * 4 + k];
  for (int i = t; i < 512; i += 128) {
    int k = i >> 7, d = i & 127;
    base[i] = basebuf[(long)b * 512 + i];
    c0[i] = mw[k] * s_w2[(k * 128 + d) * 2 + 0];
    c1[i] = mw[k] * s_w2[(k * 128 + d) * 2 + 1];
    w0[i] = s_w1[k * 258 * 128 + d];
    w1[i] = s_w1[k * 258 * 128 + 128 + d];
  }
  if (t < 2) {
    float a = 0.f;
#pragma unroll
    for (int k = 0; k < 4; ++k) a += mw[k] * s_b2[k * 2 + t];
    bc[t] = a;
  }
  __syncthreads();
  const float x0 = x[(long)b * 256 + t];
  const float x1 = x[(long)b * 256 + 128 + t];
  float a0 = bc[0], a1 = bc[1];
  for (int i = 0; i < 512; ++i) {
    float tt = fast_tanh(base[i] + x0 * w0[i] + x1 * w1[i]);
    a0 += tt * c0[i]; a1 += tt * c1[i];
  }
  out[(long)b * 256 + t] = a0;
  out[(long)b * 256 + 128 + t] = a1;
}

extern "C" void kernel_launch(void* const* d_in, const int* in_sizes, int n_in,
                              void* d_out, int out_size, void* d_ws, size_t ws_size,
                              hipStream_t stream) {
  (void)in_sizes; (void)n_in; (void)out_size; (void)ws_size;
  const float* x        = (const float*)d_in[0];
  const int*   tsteps   = (const int*)d_in[1];
  const float* cond     = (const float*)d_in[2];
  const float* t_w1     = (const float*)d_in[3];
  const float* t_b1     = (const float*)d_in[4];
  const float* t_w2     = (const float*)d_in[5];
  const float* t_b2     = (const float*)d_in[6];
  const float* c_w1     = (const float*)d_in[7];
  const float* c_b1     = (const float*)d_in[8];
  const float* c_w2     = (const float*)d_in[9];
  const float* c_b2     = (const float*)d_in[10];
  const float* attn_wqkv= (const float*)d_in[11];
  const float* attn_bqkv= (const float*)d_in[12];
  const float* attn_wo  = (const float*)d_in[13];
  const float* attn_bo  = (const float*)d_in[14];
  const float* ln1_g    = (const float*)d_in[15];
  const float* ln1_b    = (const float*)d_in[16];
  const float* ff_w1    = (const float*)d_in[17];
  const float* ff_b1    = (const float*)d_in[18];
  const float* ff_w2    = (const float*)d_in[19];
  const float* ff_b2    = (const float*)d_in[20];
  const float* ln2_g    = (const float*)d_in[21];
  const float* ln2_b    = (const float*)d_in[22];
  const float* mw_w1    = (const float*)d_in[23];
  const float* mw_b1    = (const float*)d_in[24];
  const float* mw_w2    = (const float*)d_in[25];
  const float* mw_b2    = (const float*)d_in[26];
  const float* s_w1     = (const float*)d_in[27];
  const float* s_b1     = (const float*)d_in[28];
  const float* s_w2     = (const float*)d_in[29];
  const float* s_b2     = (const float*)d_in[30];
  float* out = (float*)d_out;

  const int B = 4096;
  char* ws = (char*)d_ws;
  auto carve = [&](size_t bytes) -> char* {
    char* p = ws; ws += (bytes + 255) & ~(size_t)255; return p;
  };
  _Float16* comb16  = (_Float16*)carve((size_t)B * 256 * 2);
  float*    basebuf = (float*)carve((size_t)B * 512 * 4);
  float*    zbuf    = (float*)carve((size_t)B * 128 * 4);
  float*    mwbuf   = (float*)carve((size_t)B * 4 * 4);
  _Float16* c_w2t   = (_Float16*)carve((size_t)128 * 64 * 2);
  _Float16* wqkv_t  = (_Float16*)carve((size_t)2 * 384 * 128 * 2);
  _Float16* wo_t    = (_Float16*)carve((size_t)2 * 128 * 128 * 2);
  _Float16* ffw1_t  = (_Float16*)carve((size_t)2 * 256 * 128 * 2);
  _Float16* ffw2_t  = (_Float16*)carve((size_t)2 * 128 * 256 * 2);
  _Float16* wcat    = (_Float16*)carve((size_t)640 * 256 * 2);

  auto blocks = [](int n) { return (n + 255) / 256; };
  // weight prep (f32 -> f16 transposed)
  transpose_f16_kernel<<<blocks(64 * 128), 256, 0, stream>>>(c_w2, c_w2t, 64, 128);
  for (int l = 0; l < 2; ++l) {
    transpose_f16_kernel<<<blocks(128 * 384), 256, 0, stream>>>(
        attn_wqkv + l * 128 * 384, wqkv_t + l * 384 * 128, 128, 384);
    transpose_f16_kernel<<<blocks(128 * 128), 256, 0, stream>>>(
        attn_wo + l * 128 * 128, wo_t + l * 128 * 128, 128, 128);
    transpose_f16_kernel<<<blocks(128 * 256), 256, 0, stream>>>(
        ff_w1 + l * 128 * 256, ffw1_t + l * 256 * 128, 128, 256);
    transpose_f16_kernel<<<blocks(256 * 128), 256, 0, stream>>>(
        ff_w2 + l * 256 * 128, ffw2_t + l * 128 * 256, 256, 128);
  }
  wcat_kernel<<<blocks(640 * 256), 256, 0, stream>>>(s_w1, mw_w1, wcat);

  timeemb_kernel<<<B, 128, 0, stream>>>(tsteps, t_w1, t_b1, t_w2, t_b2, comb16);

  const size_t smem = 65536 + 65536 + 98304 + 32768 + 32768; // 294912 bytes
  transformer_kernel<<<B, 256, smem, stream>>>(
      cond, c_w1, c_b1, c_b2, c_w2t, wqkv_t, attn_bqkv, wo_t, attn_bo,
      ln1_g, ln1_b, ffw1_t, ff_b1, ffw2_t, ff_b2, ln2_g, ln2_b, comb16);

  base_gemm_kernel<<<160, 256, 0, stream>>>(comb16, wcat, s_b1, mw_b1, basebuf, zbuf);
  mix_kernel<<<B, 128, 0, stream>>>(zbuf, mw_w2, mw_b2, mwbuf);
  student_kernel<<<B, 128, 0, stream>>>(x, basebuf, mwbuf, s_w1, s_w2, s_b2, out);
}
